// DeconvDecoder_18863496364236
// MI455X (gfx1250) — matrix operand order repaired and checked by reading, not compiled, not run on hardware
//
#include <hip/hip_runtime.h>

// ---------------------------------------------------------------------------
// MI455X (gfx1250) implementation.
// Precision: bf16 inputs to V_WMMA_F32_16X16X32_BF16, f32 accumulate.
// Deconv(k=4,s=2,p=1) == 4 parity-class GEMMs with K = 4 taps * Cin.
// A-tiles streamed global->LDS with GLOBAL_LOAD_ASYNC_TO_LDS_B128 (ASYNCcnt),
// double-buffered so the copy of tile k+1 overlaps WMMA on tile k.
// ---------------------------------------------------------------------------

typedef unsigned int u32;
typedef unsigned short u16;
typedef __attribute__((ext_vector_type(16))) __bf16 v16bf;
typedef __attribute__((ext_vector_type(8)))  float  v8f;

__host__ __device__ constexpr int ilog2c(int v) {
  int r = 0; while (v > 1) { v >>= 1; ++r; } return r;
}

__device__ __forceinline__ u16 bf16_rne(float f) {
  u32 u = __float_as_uint(f);
  u32 r = u + 0x7fffu + ((u >> 16) & 1u);
  return (u16)(r >> 16);
}
__device__ __forceinline__ u32 pack_bf16(float lo, float hi) {
  return (u32)bf16_rne(lo) | ((u32)bf16_rne(hi) << 16);
}

// Low 32 bits of the generic address of a __shared__ object == wave-relative
// LDS byte address (flat aperture mapping), which is what the async-to-LDS
// VDST operand expects.
__device__ __forceinline__ u32 lds_addr_u32(const void* p) {
  return (u32)(size_t)p;
}

// GVS-mode async copy: LDS[loff] <- MEM[saddr + goff], 16 bytes, ASYNCcnt++.
__device__ __forceinline__ void async_load_b128(u32 loff, u32 goff, const void* sbase) {
  asm volatile("global_load_async_to_lds_b128 %0, %1, %2"
               :: "v"(loff), "v"(goff), "s"(sbase) : "memory");
}
__device__ __forceinline__ void wait_async_le2() {
  asm volatile("s_wait_asynccnt 0x2" ::: "memory");
}
__device__ __forceinline__ void wait_async_0() {
  asm volatile("s_wait_asynccnt 0x0" ::: "memory");
}

union FragU { uint4 q[2]; v16bf v; };

// A-matrix 16x32 bf16 fragment from row-major LDS tile (rows x 32 bf16, 16 words/row).
// lane<16: K0..15 (words 0..7); lane>=16: K16..31 (words 8..15).
__device__ __forceinline__ v16bf load_a_frag(const u32* ldsA, int row, int lane) {
  const uint4* p = (const uint4*)(ldsA + row * 16 + ((lane >> 4) << 3));
  FragU u; u.q[0] = p[0]; u.q[1] = p[1];
  return u.v;
}
// B fragment pre-swizzled in LDS: 32 bytes contiguous per lane.
__device__ __forceinline__ v16bf load_b_frag(const u32* fragbase, int lane) {
  const uint4* p = (const uint4*)(fragbase + lane * 8);
  FragU u; u.q[0] = p[0]; u.q[1] = p[1];
  return u.v;
}

// ---------------------------------------------------------------------------
// Zero the 64B OOB pads at the tail of each deconv input buffer (per launch).
// ---------------------------------------------------------------------------
__global__ void init_pads_kernel(u16* p1, u16* p2, u16* p3, u16* p4) {
  int t = threadIdx.x;
  if (t < 32) { p1[t] = 0; p2[t] = 0; p3[t] = 0; p4[t] = 0; }
}

// ---------------------------------------------------------------------------
// Kernel 1: latent matmul + 10 switched-MLP layers, fused. One wave per row.
// Lane holds features d = lane + 32*j, j=0..5 (D=192). Output: relu -> bf16.
// ---------------------------------------------------------------------------
__global__ __launch_bounds__(256) void latent_switch_kernel(
    const float* __restrict__ z, const int* __restrict__ ys,
    const float* __restrict__ w_lat, const float* __restrict__ b_lat,
    const float* __restrict__ w_in, const float* __restrict__ b_in,
    const float* __restrict__ w_out, const float* __restrict__ b_out,
    u16* __restrict__ act0) {
  const int lane = threadIdx.x & 31;
  const int row  = blockIdx.x * 8 + (threadIdx.x >> 5);

  float zv[10];
  #pragma unroll
  for (int l = 0; l < 10; ++l) zv[l] = z[row * 10 + l];

  float outv[6];
  #pragma unroll
  for (int j = 0; j < 6; ++j) {
    int d = lane + 32 * j;
    float acc = b_lat[d];
    #pragma unroll
    for (int l = 0; l < 10; ++l) acc += zv[l] * w_lat[l * 192 + d];
    outv[j] = acc;
  }

  for (int i = 0; i < 10; ++i) {
    int k = ys[i * 4096 + row];
    const float* Wi = w_in  + (size_t)((i * 10 + k) * 192) * 10;
    const float* Wo = w_out + (size_t)((i * 10 + k) * 10) * 192;
    const float* Bi = b_in  + (i * 10 + k) * 10;
    const float* Bo = b_out + (size_t)(i * 10 + k) * 192;

    float r[6];
    #pragma unroll
    for (int j = 0; j < 6; ++j) r[j] = fmaxf(outv[j], 0.f);

    float h[10];
    #pragma unroll
    for (int s = 0; s < 10; ++s) {
      float p = 0.f;
      #pragma unroll
      for (int j = 0; j < 6; ++j) p += r[j] * Wi[(lane + 32 * j) * 10 + s];
      #pragma unroll
      for (int off = 16; off > 0; off >>= 1) p += __shfl_xor(p, off, 32);
      h[s] = p + Bi[s];
    }
    #pragma unroll
    for (int j = 0; j < 6; ++j) {
      int d = lane + 32 * j;
      float acc = Bo[d];
      #pragma unroll
      for (int s = 0; s < 10; ++s) acc += h[s] * Wo[s * 192 + d];
      outv[j] += acc;
    }
  }
  #pragma unroll
  for (int j = 0; j < 6; ++j) {
    int d = lane + 32 * j;
    act0[(size_t)row * 192 + d] = bf16_rne(fmaxf(outv[j], 0.f));
  }
}

// ---------------------------------------------------------------------------
// Kernel 2: FC GEMM [4096,192]x[192,1024] + bias + relu, WMMA bf16.
// Block tile 128x64, 8 waves (2M x 4N), K = 6 blocks of 32.
// A-tiles double-buffered via async-to-LDS. Output scattered to NHWC
// act1[B][4*4][64] bf16 for the deconv chain.
// ---------------------------------------------------------------------------
__global__ __launch_bounds__(256) void fc_kernel(
    const u16* __restrict__ act0, const float* __restrict__ w_fc,
    const float* __restrict__ b_fc, u16* __restrict__ act1) {
  __shared__ u32 ldsB[6 * 4 * 256];     // 24 KB
  __shared__ u32 ldsA[2 * 128 * 16];    // 2 x 8 KB double buffer

  const int tid = threadIdx.x, lane = tid & 31, wid = tid >> 5;
  const int mbase = blockIdx.x * 128;
  const int nblk  = blockIdx.y * 64;
  const u32 ldsA0 = lds_addr_u32(ldsA);

  // Stage B (w_fc f32 -> bf16) once, pre-swizzled to WMMA B 32x16 lane layout.
  for (int idx = tid; idx < 6 * 4 * 256; idx += 256) {
    int word = idx & 7, ln = (idx >> 3) & 31;
    int nt = (idx >> 8) & 3, kb = idx >> 10;
    int n = nblk + nt * 16 + (ln & 15);
    int k = kb * 32 + ((ln >> 4) << 4) + word * 2;
    ldsB[idx] = pack_bf16(w_fc[(size_t)k * 1024 + n], w_fc[(size_t)(k + 1) * 1024 + n]);
  }

  auto issue_tile = [&](int kb, int bufsel) {
    #pragma unroll
    for (int it = 0; it < 2; ++it) {           // wave issues exactly 2 async ops
      int idx = tid + it * 256;
      int row = idx >> 2, q = idx & 3;
      u32 goff = (u32)((((mbase + row) * 192) + kb * 32) * 2 + q * 16);
      u32 loff = ldsA0 + (u32)(bufsel * 8192 + row * 64 + q * 16);
      async_load_b128(loff, goff, act0);
    }
  };

  const int nt = wid & 3, wm = (wid >> 2) * 64;
  v8f acc[4];
  #pragma unroll
  for (int f = 0; f < 4; ++f)
    #pragma unroll
    for (int j = 0; j < 8; ++j) acc[f][j] = 0.f;

  issue_tile(0, 0);
  for (int kb = 0; kb < 6; ++kb) {
    __syncthreads();                           // buf (kb+1)&1 free to overwrite
    if (kb + 1 < 6) { issue_tile(kb + 1, (kb + 1) & 1); wait_async_le2(); }
    else            { wait_async_0(); }
    __syncthreads();                           // tile kb resident for all waves
    const u32* bufA = ldsA + (kb & 1) * 2048;
    v16bf bfrag = load_b_frag(ldsB + (kb * 4 + nt) * 256, lane);
    #pragma unroll
    for (int f = 0; f < 4; ++f) {
      v16bf afrag = load_a_frag(bufA, wm + f * 16 + (lane & 15), lane);
      acc[f] = __builtin_amdgcn_wmma_f32_16x16x32_bf16(
          false, afrag, false, bfrag, (short)0, acc[f], false, false);
    }
  }

  const int ncol = nblk + nt * 16 + (lane & 15);
  const float bias = b_fc[ncol];
  const int c = ncol >> 4, p = ncol & 15;   // fc col n -> (channel, pixel)
  #pragma unroll
  for (int f = 0; f < 4; ++f)
    #pragma unroll
    for (int j = 0; j < 8; ++j) {
      int m = mbase + wm + f * 16 + j + ((lane >> 4) << 3);
      float v = fmaxf(acc[f][j] + bias, 0.f);
      act1[((size_t)m * 16 + p) * 64 + c] = bf16_rne(v);
    }
}

// ---------------------------------------------------------------------------
// Deconv (ConvTranspose2d k=4,s=2,p=1) as implicit GEMM per parity class.
// Input NHWC bf16 [B,H,W,CIN] (+64B zero pad at end for OOB taps);
// output NHWC bf16 [B,2H,2W,RCOUT] (+relu), or (LAST) NCHW f32, no relu.
// K ordered tap-major: k = tap*CIN + ci. Block tile M=128 x N=COUTP, 8 waves.
// grid.y = parity class (py,px). A-tiles double-buffered via async-to-LDS.
// ---------------------------------------------------------------------------
template<int CIN, int COUTP, int RCOUT, int H, int W, bool LAST>
__global__ __launch_bounds__(256) void deconv_kernel(
    const u16* __restrict__ in, const float* __restrict__ w,
    const float* __restrict__ bias, u16* __restrict__ out_bf,
    float* __restrict__ out_f32) {
  constexpr int KB = (4 * CIN) / 32;
  constexpr int NT = COUTP / 16;
  constexpr int LOGW = ilog2c(W), LOGH = ilog2c(H);
  constexpr u32 PADOFF = (u32)((size_t)4096 * H * W * CIN * 2);  // zero pad bytes

  __shared__ u32 ldsB[KB * NT * 256];
  __shared__ u32 ldsA[2 * 128 * 16];

  const int tid = threadIdx.x, lane = tid & 31, wid = tid >> 5;
  const int py = blockIdx.y >> 1, px = blockIdx.y & 1;
  const int mbase = blockIdx.x * 128;
  const u32 ldsA0 = lds_addr_u32(ldsA);

  // Stage all weights once, pre-swizzled. tap (ay,ax): ay=0 -> iy=ty; ay=1 ->
  // iy=ty-1 (py=0) / ty+1 (py=1). ky: py==0 ? (ay?3:1) : (ay?0:2).
  for (int idx = tid; idx < KB * NT * 256; idx += 256) {
    int word = idx & 7, ln = (idx >> 3) & 31;
    int nt_ = (idx >> 8) % NT, kb = idx / (256 * NT);
    int co = nt_ * 16 + (ln & 15);
    int k  = kb * 32 + ((ln >> 4) << 4) + word * 2;
    u32 pk = 0;
    if (co < RCOUT) {
      int tap = k / CIN, ci = k % CIN;
      int ay = tap >> 1, ax = tap & 1;
      int ky = py ? (ay ? 0 : 2) : (ay ? 3 : 1);
      int kx = px ? (ax ? 0 : 2) : (ax ? 3 : 1);
      float lo = w[(((size_t)ci       * RCOUT + co) * 4 + ky) * 4 + kx];
      float hi = w[(((size_t)(ci + 1) * RCOUT + co) * 4 + ky) * 4 + kx];
      pk = pack_bf16(lo, hi);
    }
    ldsB[idx] = pk;
  }

  auto issue_tile = [&](int kb, int bufsel) {
    const int tap = (kb * 32) / CIN;
    const int cibase = (kb * 32) % CIN;
    const int dy = (tap >> 1) ? (py ? 1 : -1) : 0;
    const int dx = (tap &  1) ? (px ? 1 : -1) : 0;
    #pragma unroll
    for (int it = 0; it < 2; ++it) {           // wave issues exactly 2 async ops
      int idx = tid + it * 256;
      int row = idx >> 2, q = idx & 3;
      int m = mbase + row;
      int tx = m & (W - 1), ty = (m >> LOGW) & (H - 1), b = m >> (LOGW + LOGH);
      int iy = ty + dy, ix = tx + dx;
      u32 goff;
      if ((unsigned)iy < (unsigned)H && (unsigned)ix < (unsigned)W)
        goff = (u32)((((b * H + iy) * W + ix) * CIN + cibase) * 2 + q * 16);
      else
        goff = PADOFF + (u32)(q * 16);         // reads zeros from the pad
      u32 loff = ldsA0 + (u32)(bufsel * 8192 + row * 64 + q * 16);
      async_load_b128(loff, goff, in);
    }
  };

  const int nt = wid % NT;
  const int wm = (wid / NT) * (16 * NT);
  v8f acc[NT];
  #pragma unroll
  for (int f = 0; f < NT; ++f)
    #pragma unroll
    for (int j = 0; j < 8; ++j) acc[f][j] = 0.f;

  issue_tile(0, 0);
  for (int kb = 0; kb < KB; ++kb) {
    __syncthreads();                           // buf (kb+1)&1 free to overwrite
    if (kb + 1 < KB) { issue_tile(kb + 1, (kb + 1) & 1); wait_async_le2(); }
    else             { wait_async_0(); }
    __syncthreads();                           // tile kb resident for all waves
    const u32* bufA = ldsA + (kb & 1) * 2048;
    v16bf bfrag = load_b_frag(ldsB + (kb * NT + nt) * 256, lane);
    #pragma unroll
    for (int f = 0; f < NT; ++f) {
      v16bf afrag = load_a_frag(bufA, wm + f * 16 + (lane & 15), lane);
      acc[f] = __builtin_amdgcn_wmma_f32_16x16x32_bf16(
          false, afrag, false, bfrag, (short)0, acc[f], false, false);
    }
  }

  const int co = nt * 16 + (lane & 15);
  const float bv = (co < RCOUT) ? bias[co] : 0.f;
  #pragma unroll
  for (int f = 0; f < NT; ++f) {
    #pragma unroll
    for (int j = 0; j < 8; ++j) {
      int m = mbase + wm + f * 16 + j + ((lane >> 4) << 3);
      int tx = m & (W - 1), ty = (m >> LOGW) & (H - 1), b = m >> (LOGW + LOGH);
      int oy = 2 * ty + py, ox = 2 * tx + px;
      float v = acc[f][j] + bv;
      if (!LAST) {
        v = fmaxf(v, 0.f);
        out_bf[((size_t)((b * (2 * H) + oy) * (2 * W)) + ox) * RCOUT + co] = bf16_rne(v);
      } else if (co < RCOUT) {
        out_f32[(((size_t)b * RCOUT + co) * (2 * H) + oy) * (2 * W) + ox] = v;
      }
    }
  }
}

// ---------------------------------------------------------------------------
extern "C" void kernel_launch(void* const* d_in, const int* in_sizes, int n_in,
                              void* d_out, int out_size, void* d_ws, size_t ws_size,
                              hipStream_t stream) {
  (void)in_sizes; (void)n_in; (void)out_size; (void)ws_size;
  const float* z        = (const float*)d_in[0];
  const int*   ys       = (const int*)  d_in[1];
  const float* w_lat    = (const float*)d_in[2];
  const float* b_lat    = (const float*)d_in[3];
  const float* sw_w_in  = (const float*)d_in[4];
  const float* sw_b_in  = (const float*)d_in[5];
  const float* sw_w_out = (const float*)d_in[6];
  const float* sw_b_out = (const float*)d_in[7];
  const float* w_fc     = (const float*)d_in[8];
  const float* b_fc     = (const float*)d_in[9];
  const float* w_d1     = (const float*)d_in[10];
  const float* b_d1     = (const float*)d_in[11];
  const float* w_d2     = (const float*)d_in[12];
  const float* b_d2     = (const float*)d_in[13];
  const float* w_d3     = (const float*)d_in[14];
  const float* b_d3     = (const float*)d_in[15];
  const float* w_d4     = (const float*)d_in[16];
  const float* b_d4     = (const float*)d_in[17];

  char* ws = (char*)d_ws;
  auto alignup = [](size_t x) { return (x + 255) & ~(size_t)255; };
  const size_t n1 = (size_t)4096 * 16 * 64;     // deconv1 input elems
  const size_t n2 = (size_t)4096 * 64 * 64;     // deconv2 input elems
  const size_t n3 = (size_t)4096 * 256 * 32;    // deconv3 input elems
  const size_t n4 = (size_t)4096 * 1024 * 32;   // deconv4 input elems
  size_t off = 0;
  u16* act0 = (u16*)(ws + off); off = alignup(off + (size_t)4096 * 192 * 2);
  u16* act1 = (u16*)(ws + off); off = alignup(off + (n1 + 64) * 2);
  u16* act2 = (u16*)(ws + off); off = alignup(off + (n2 + 64) * 2);
  u16* act3 = (u16*)(ws + off); off = alignup(off + (n3 + 64) * 2);
  u16* act4 = (u16*)(ws + off); off = alignup(off + (n4 + 64) * 2);

  init_pads_kernel<<<1, 32, 0, stream>>>(act1 + n1, act2 + n2, act3 + n3, act4 + n4);

  latent_switch_kernel<<<512, 256, 0, stream>>>(
      z, ys, w_lat, b_lat, sw_w_in, sw_b_in, sw_w_out, sw_b_out, act0);

  fc_kernel<<<dim3(32, 16), 256, 0, stream>>>(act0, w_fc, b_fc, act1);

  deconv_kernel<64, 64, 64,  4,  4, false><<<dim3(  512, 4), 256, 0, stream>>>(
      act1, w_d1, b_d1, act2, nullptr);
  deconv_kernel<64, 32, 32,  8,  8, false><<<dim3( 2048, 4), 256, 0, stream>>>(
      act2, w_d2, b_d2, act3, nullptr);
  deconv_kernel<32, 32, 32, 16, 16, false><<<dim3( 8192, 4), 256, 0, stream>>>(
      act3, w_d3, b_d3, act4, nullptr);
  deconv_kernel<32, 16,  3, 32, 32, true ><<<dim3(32768, 4), 256, 0, stream>>>(
      act4, w_d4, b_d4, nullptr, (float*)d_out);
}
